// EdgeUpdate_14164802142345
// MI455X (gfx1250) — matrix operand order repaired and checked by reading, not compiled, not run on hardware
//
#include <hip/hip_runtime.h>
#include <hip/hip_bf16.h>
#include <cstdint>

// ---------------------------------------------------------------------------
// EdgeUpdate (AlphaFold-style pair stack) for MI455X / gfx1250, wave32 + WMMA.
// All GEMMs run on v_wmma_f32_16x16x32_f16 (f16 in, f32 accumulate).
// ---------------------------------------------------------------------------

typedef __attribute__((ext_vector_type(16))) _Float16 v16h;
typedef __attribute__((ext_vector_type(8)))  _Float16 v8h;
typedef __attribute__((ext_vector_type(4)))  _Float16 v4h;
typedef __attribute__((ext_vector_type(8)))  float    v8f;

#define NRES 512
#define CP   128
#define NPAIR (NRES * NRES)        // 262144
#define FK   160                   // feats padded 148 -> 160
#define LDT  48                    // LDS tile row stride (halves); 96B = 6*16B aligned

// ============================== WMMA fragments ==============================

__device__ __forceinline__ v16h frag_a_lds(const _Float16* tile, int m, int lane) {
    // A 16x32 f16: lane%16 = M row; halves 0..7 = K kb..kb+7, 8..15 = K kb+16..kb+23
    const _Float16* p = tile + m * LDT;
    const int kb = (lane >> 4) << 3;                 // 0 or 8
    v8h lo = *(const v8h*)(p + kb);
    v8h hi = *(const v8h*)(p + kb + 16);
    v16h f;
#pragma unroll
    for (int t = 0; t < 8; ++t) { f[t] = lo[t]; f[t + 8] = hi[t]; }
    return f;
}

__device__ __forceinline__ v16h frag_b_lds(const _Float16* tile, int n, int lane) {
    // B 32x16 f16 stored transposed in LDS: tile[n][k]; lane%16 = N col,
    // halves 0..7 = K kb..kb+7, 8..15 = K kb+16..kb+23, kb = 8*(lane>=16)
    const _Float16* p = tile + n * LDT + ((lane >> 4) << 3);
    v8h lo = *(const v8h*)(p);
    v8h hi = *(const v8h*)(p + 16);
    v16h f;
#pragma unroll
    for (int t = 0; t < 8; ++t) { f[t] = lo[t]; f[t + 8] = hi[t]; }
    return f;
}

// ============================ Generic WMMA GEMM =============================
// C[M,N] = act(A[M,K](f16,lda) @ W[K,N](f32,ldw) + bias)
// 256 threads = 8 waves; block tile 128(M) x 128(N); K staged 32 at a time.
// lda must be a multiple of 32 (buffers are zero-padded past K).

#define ACT_NONE 0
#define ACT_RELU 1
#define ACT_SIG  2

template <int ACT, bool F16OUT>
__global__ void __launch_bounds__(256)
gemm_kernel(const _Float16* __restrict__ A, int lda,
            const float* __restrict__ W, int ldw,
            const float* __restrict__ bias,
            void* __restrict__ Cout, int ldc, int K) {
    __shared__ _Float16 aT[128 * LDT];
    __shared__ _Float16 wT[128 * LDT];

    const int t    = threadIdx.x;
    const int wave = t >> 5;
    const int lane = t & 31;
    const int m0   = blockIdx.x * 128;
    const int n0   = blockIdx.y * 128;

    v8f acc[8];
#pragma unroll
    for (int i = 0; i < 8; ++i) acc[i] = (v8f)(0.0f);

    for (int k0 = 0; k0 < K; k0 += 32) {
        // --- stage A tile: 128 rows x 32 halves (vector, lda-padded past K) ---
        {
            const int row = t >> 1;
            const int seg = (t & 1) << 4;
            const _Float16* src = A + (size_t)(m0 + row) * lda + k0 + seg;
            v8h x0 = *(const v8h*)(src);
            v8h x1 = *(const v8h*)(src + 8);
            _Float16* dst = aT + row * LDT + seg;
            *(v8h*)(dst)     = x0;
            *(v8h*)(dst + 8) = x1;
        }
        // --- stage W tile transposed + f32->f16 convert, zero pad k>=K ---
#pragma unroll
        for (int idx = t; idx < 32 * 128; idx += 256) {
            const int nn = idx & 127;
            const int kk = idx >> 7;
            const int k  = k0 + kk;
            float v = (k < K) ? W[(size_t)k * ldw + n0 + nn] : 0.0f;
            wT[nn * LDT + kk] = (_Float16)v;
        }
        __syncthreads();

        v16h fa = frag_a_lds(aT, (wave << 4) + (lane & 15), lane);
#pragma unroll
        for (int nt = 0; nt < 8; ++nt) {
            v16h fb = frag_b_lds(wT, (nt << 4) + (lane & 15), lane);
            acc[nt] = __builtin_amdgcn_wmma_f32_16x16x32_f16(
                false, fa, false, fb, (short)0, acc[nt], false, false);
        }
        __syncthreads();
    }

    // --- epilogue ---
    const int colb  = n0 + (lane & 15);
    const int rbase = m0 + (wave << 4) + ((lane >> 4) << 3);
#pragma unroll
    for (int nt = 0; nt < 8; ++nt) {
        const int col = colb + (nt << 4);
        const float bv = bias ? bias[col] : 0.0f;
#pragma unroll
        for (int v = 0; v < 8; ++v) {
            float x = acc[nt][v] + bv;
            if (ACT == ACT_RELU) x = fmaxf(x, 0.0f);
            if (ACT == ACT_SIG)  x = 1.0f / (1.0f + __expf(-x));
            const size_t o = (size_t)(rbase + v) * ldc + col;
            if (F16OUT) ((_Float16*)Cout)[o] = (_Float16)x;
            else        ((float*)Cout)[o]    = x;
        }
    }
}

// ===================== Triangle einsum (batched per channel) ================
// X[i,j,c] = sum_k at[c,i,k] * bt[c,j,k]   (A @ B^T, per channel c)
// grid = (4, 4, 128); block tile 128x128; K = 512.

__global__ void __launch_bounds__(256)
tri_einsum_kernel(const _Float16* __restrict__ AT, const _Float16* __restrict__ BT,
                  float* __restrict__ X) {
    __shared__ _Float16 aT[128 * LDT];
    __shared__ _Float16 bT[128 * LDT];

    const int t    = threadIdx.x;
    const int wave = t >> 5;
    const int lane = t & 31;
    const int i0   = blockIdx.x * 128;
    const int j0   = blockIdx.y * 128;
    const int c    = blockIdx.z;

    const _Float16* Ab = AT + (size_t)c * NPAIR;
    const _Float16* Bb = BT + (size_t)c * NPAIR;

    v8f acc[8];
#pragma unroll
    for (int i = 0; i < 8; ++i) acc[i] = (v8f)(0.0f);

    for (int k0 = 0; k0 < NRES; k0 += 32) {
        const int row = t >> 1;
        const int seg = (t & 1) << 4;
        {
            const _Float16* src = Ab + (size_t)(i0 + row) * NRES + k0 + seg;
            v8h x0 = *(const v8h*)(src);
            v8h x1 = *(const v8h*)(src + 8);
            *(v8h*)(aT + row * LDT + seg)     = x0;
            *(v8h*)(aT + row * LDT + seg + 8) = x1;
        }
        {
            const _Float16* src = Bb + (size_t)(j0 + row) * NRES + k0 + seg;
            v8h x0 = *(const v8h*)(src);
            v8h x1 = *(const v8h*)(src + 8);
            *(v8h*)(bT + row * LDT + seg)     = x0;
            *(v8h*)(bT + row * LDT + seg + 8) = x1;
        }
        __syncthreads();

        v16h fa = frag_a_lds(aT, (wave << 4) + (lane & 15), lane);
#pragma unroll
        for (int nt = 0; nt < 8; ++nt) {
            v16h fb = frag_b_lds(bT, (nt << 4) + (lane & 15), lane);
            acc[nt] = __builtin_amdgcn_wmma_f32_16x16x32_f16(
                false, fa, false, fb, (short)0, acc[nt], false, false);
        }
        __syncthreads();
    }

    const int ibase = i0 + (wave << 4) + ((lane >> 4) << 3);
    const int jbase = j0 + (lane & 15);
#pragma unroll
    for (int nt = 0; nt < 8; ++nt) {
        const int j = jbase + (nt << 4);
#pragma unroll
        for (int v = 0; v < 8; ++v) {
            const int i = ibase + v;
            X[((size_t)i * NRES + j) * CP + c] = acc[nt][v];
        }
    }
}

// ================================ LayerNorm =================================
// One wave per 128-channel row, f32 in -> f16 out.

__global__ void __launch_bounds__(256)
ln_kernel(const float* __restrict__ X, const float* __restrict__ g,
          const float* __restrict__ b, _Float16* __restrict__ Y) {
    const int wave = threadIdx.x >> 5, lane = threadIdx.x & 31;
    const size_t row = (size_t)blockIdx.x * 8 + wave;
    const float4 xv = *(const float4*)(X + row * CP + lane * 4);
    float s = xv.x + xv.y + xv.z + xv.w;
    float q = xv.x * xv.x + xv.y * xv.y + xv.z * xv.z + xv.w * xv.w;
#pragma unroll
    for (int off = 16; off > 0; off >>= 1) {
        s += __shfl_xor(s, off, 32);
        q += __shfl_xor(q, off, 32);
    }
    const float mean = s * (1.0f / CP);
    const float rstd = rsqrtf(q * (1.0f / CP) - mean * mean + 1e-5f);
    const float4 gv = *(const float4*)(g + lane * 4);
    const float4 bv = *(const float4*)(b + lane * 4);
    v4h o;
    o[0] = (_Float16)((xv.x - mean) * rstd * gv.x + bv.x);
    o[1] = (_Float16)((xv.y - mean) * rstd * gv.y + bv.y);
    o[2] = (_Float16)((xv.z - mean) * rstd * gv.z + bv.z);
    o[3] = (_Float16)((xv.w - mean) * rstd * gv.w + bv.w);
    *(v4h*)(Y + row * CP + lane * 4) = o;
}

// ============================= Pair features ================================
// feats[m, 0:128]  = LN(sb_i * sb_j) (saln),  feats[m, 128:148] = RBF(d_ij),
// feats[m, 148:160] = 0.   One wave per pair m = i*512 + j.

__global__ void __launch_bounds__(256)
pairfeat_kernel(const float* __restrict__ SB, const float* __restrict__ trans,
                const float* __restrict__ g, const float* __restrict__ b,
                _Float16* __restrict__ F) {
    const int wave = threadIdx.x >> 5, lane = threadIdx.x & 31;
    const size_t m = (size_t)blockIdx.x * 8 + wave;
    const int i = (int)(m >> 9), j = (int)(m & (NRES - 1));

    const float4 ai = *(const float4*)(SB + (size_t)i * CP + lane * 4);
    const float4 aj = *(const float4*)(SB + (size_t)j * CP + lane * 4);
    float4 x;
    x.x = ai.x * aj.x; x.y = ai.y * aj.y; x.z = ai.z * aj.z; x.w = ai.w * aj.w;
    float s = x.x + x.y + x.z + x.w;
    float q = x.x * x.x + x.y * x.y + x.z * x.z + x.w * x.w;
#pragma unroll
    for (int off = 16; off > 0; off >>= 1) {
        s += __shfl_xor(s, off, 32);
        q += __shfl_xor(q, off, 32);
    }
    const float mean = s * (1.0f / CP);
    const float rstd = rsqrtf(q * (1.0f / CP) - mean * mean + 1e-5f);
    const float4 gv = *(const float4*)(g + lane * 4);
    const float4 bv = *(const float4*)(b + lane * 4);
    v4h o;
    o[0] = (_Float16)((x.x - mean) * rstd * gv.x + bv.x);
    o[1] = (_Float16)((x.y - mean) * rstd * gv.y + bv.y);
    o[2] = (_Float16)((x.z - mean) * rstd * gv.z + bv.z);
    o[3] = (_Float16)((x.w - mean) * rstd * gv.w + bv.w);
    *(v4h*)(F + m * FK + lane * 4) = o;

    // RBF channels: lane t -> channel 128+t (t<20 valid, rest zero-pad)
    const float dx = trans[i * 3 + 0] - trans[j * 3 + 0];
    const float dy = trans[i * 3 + 1] - trans[j * 3 + 1];
    const float dz = trans[i * 3 + 2] - trans[j * 3 + 2];
    const float d  = sqrtf(dx * dx + dy * dy + dz * dz);
    const float mu = 0.001f + (float)lane * (19.999f / 19.0f);
    const float sg = 19.999f / 20.0f;
    const float r  = (d - mu) / sg;
    const float val = (lane < 20) ? __expf(-r * r) : 0.0f;
    F[m * FK + CP + lane] = (_Float16)val;
}

// ============================ Elementwise fusions ===========================

// P1 = p + (i==j ? s_i : upd)        (symmetric MLP output; diag gets s)
__global__ void __launch_bounds__(256)
fuse_bias_kernel(const float* __restrict__ P, const _Float16* __restrict__ U,
                 const float* __restrict__ S, float* __restrict__ P1) {
    const size_t gid = (size_t)blockIdx.x * 256 + threadIdx.x;  // 4 channels each
    const size_t m = gid >> 5;
    const int cc = (int)(gid & 31) << 2;
    const int i = (int)(m >> 9), j = (int)(m & (NRES - 1));
    float4 pv = *(const float4*)(P + m * CP + cc);
    if (i == j) {
        const float4 sv = *(const float4*)(S + (size_t)i * CP + cc);
        pv.x += sv.x; pv.y += sv.y; pv.z += sv.z; pv.w += sv.w;
    } else {
        const v4h uv = *(const v4h*)(U + m * CP + cc);
        pv.x += (float)uv[0]; pv.y += (float)uv[1];
        pv.z += (float)uv[2]; pv.w += (float)uv[3];
    }
    *(float4*)(P1 + m * CP + cc) = pv;
}

// at[c][i][k] = mask[src] * gate[src,c] * proj[src,c];  src = outgoing ? (i,k) : (k,i)
template <bool OUTGOING>
__global__ void __launch_bounds__(256)
fuse_mul_tr_kernel(const _Float16* __restrict__ G, const _Float16* __restrict__ Pj,
                   const float* __restrict__ mask, _Float16* __restrict__ T) {
    const size_t o = (size_t)blockIdx.x * 256 + threadIdx.x;  // c*NPAIR + i*512 + k
    const int k = (int)(o & (NRES - 1));
    const int i = (int)((o >> 9) & (NRES - 1));
    const int c = (int)(o >> 18);
    const size_t src = OUTGOING ? ((size_t)i * NRES + k) : ((size_t)k * NRES + i);
    const float v = mask[src] * (float)G[src * CP + c] * (float)Pj[src * CP + c];
    T[o] = (_Float16)v;
}

// P1 += g * zout
__global__ void __launch_bounds__(256)
fuse_gate_kernel(const _Float16* __restrict__ G, const _Float16* __restrict__ Z,
                 float* __restrict__ P1) {
    const size_t gid = (size_t)blockIdx.x * 256 + threadIdx.x;
    const size_t o = gid << 2;
    float4 pv = *(const float4*)(P1 + o);
    const v4h gv = *(const v4h*)(G + o);
    const v4h zv = *(const v4h*)(Z + o);
    pv.x += (float)gv[0] * (float)zv[0];
    pv.y += (float)gv[1] * (float)zv[1];
    pv.z += (float)gv[2] * (float)zv[2];
    pv.w += (float)gv[3] * (float)zv[3];
    *(float4*)(P1 + o) = pv;
}

// out = (P1 + t*mask) * mask
__global__ void __launch_bounds__(256)
final_kernel(const float* __restrict__ P1, const _Float16* __restrict__ T,
             const float* __restrict__ mask, float* __restrict__ Out) {
    const size_t gid = (size_t)blockIdx.x * 256 + threadIdx.x;
    const size_t o = gid << 2;
    const float mk = mask[gid >> 5];
    float4 pv = *(const float4*)(P1 + o);
    const v4h tv = *(const v4h*)(T + o);
    float4 r;
    r.x = (pv.x + (float)tv[0] * mk) * mk;
    r.y = (pv.y + (float)tv[1] * mk) * mk;
    r.z = (pv.z + (float)tv[2] * mk) * mk;
    r.w = (pv.w + (float)tv[3] * mk) * mk;
    *(float4*)(Out + o) = r;
}

__global__ void __launch_bounds__(256)
cvt_f32_f16_kernel(const float* __restrict__ X, _Float16* __restrict__ Y) {
    const size_t o = (size_t)blockIdx.x * 256 + threadIdx.x;
    Y[o] = (_Float16)X[o];
}

// ================================== Host ====================================

// d_in layout (setup_inputs dict order, leaves flattened recursively):
enum {
    IN_S = 0, IN_P, IN_TRANS, IN_PMASK,
    P_SA_W, P_SA_B, P_SALN_G, P_SALN_B,
    P_FA1_W, P_FA1_B, P_FA2_W, P_FA2_B, P_FA3_W, P_FA3_B,
    TMO_BASE = 14,                // 16 entries
    TMI_BASE = TMO_BASE + 16,     // 16 entries
    P_PT_LN_G = TMI_BASE + 16, P_PT_LN_B, P_PT1_W, P_PT1_B, P_PT2_W, P_PT2_B
};
// tri block offsets within TM*_BASE:
enum { T_LNI_G = 0, T_LNI_B, T_AG_W, T_AG_B, T_AP_W, T_AP_B,
       T_BG_W, T_BG_B, T_BP_W, T_BP_B, T_G_W, T_G_B,
       T_LNO_G, T_LNO_B, T_Z_W, T_Z_B };

template <int ACT, bool F16OUT>
static inline void launch_gemm(const _Float16* A, int lda, const float* W, int ldw,
                               const float* bias, void* C, int ldc,
                               int M, int N, int K, hipStream_t st) {
    dim3 grid(M / 128, N / 128);
    gemm_kernel<ACT, F16OUT><<<grid, 256, 0, st>>>(A, lda, W, ldw, bias, C, ldc, K);
}

static void run_tri_mul(void* const* d_in, int base, bool outgoing,
                        float* P1, _Float16* ZL, _Float16* T0, _Float16* T1,
                        _Float16* T2, _Float16* T3, float* XF,
                        const float* pmask, hipStream_t st) {
    const float* ln_in_g  = (const float*)d_in[base + T_LNI_G];
    const float* ln_in_b  = (const float*)d_in[base + T_LNI_B];
    const float* ln_out_g = (const float*)d_in[base + T_LNO_G];
    const float* ln_out_b = (const float*)d_in[base + T_LNO_B];

    const int EW_BLK = (NPAIR * CP) / (256 * 4);   // elementwise 4/thread
    const int TR_BLK = (NPAIR * CP) / 256;         // transpose 1/thread

    // z_ln
    ln_kernel<<<NPAIR / 8, 256, 0, st>>>(P1, ln_in_g, ln_in_b, ZL);
    // a = mask * sigmoid(z@ag)*(z@ap)  -> channel-major T2
    launch_gemm<ACT_SIG,  true>(ZL, CP, (const float*)d_in[base + T_AG_W], CP,
                                (const float*)d_in[base + T_AG_B], T0, CP, NPAIR, CP, CP, st);
    launch_gemm<ACT_NONE, true>(ZL, CP, (const float*)d_in[base + T_AP_W], CP,
                                (const float*)d_in[base + T_AP_B], T1, CP, NPAIR, CP, CP, st);
    if (outgoing) fuse_mul_tr_kernel<true ><<<TR_BLK, 256, 0, st>>>(T0, T1, pmask, T2);
    else          fuse_mul_tr_kernel<false><<<TR_BLK, 256, 0, st>>>(T0, T1, pmask, T2);
    // b -> channel-major T3
    launch_gemm<ACT_SIG,  true>(ZL, CP, (const float*)d_in[base + T_BG_W], CP,
                                (const float*)d_in[base + T_BG_B], T0, CP, NPAIR, CP, CP, st);
    launch_gemm<ACT_NONE, true>(ZL, CP, (const float*)d_in[base + T_BP_W], CP,
                                (const float*)d_in[base + T_BP_B], T1, CP, NPAIR, CP, CP, st);
    if (outgoing) fuse_mul_tr_kernel<true ><<<TR_BLK, 256, 0, st>>>(T0, T1, pmask, T3);
    else          fuse_mul_tr_kernel<false><<<TR_BLK, 256, 0, st>>>(T0, T1, pmask, T3);
    // x[i,j,c] = sum_k at[c,i,k]*bt[c,j,k]
    tri_einsum_kernel<<<dim3(4, 4, CP), 256, 0, st>>>(T2, T3, XF);
    // x = LN_out(x) -> T0 (f16)
    ln_kernel<<<NPAIR / 8, 256, 0, st>>>(XF, ln_out_g, ln_out_b, T0);
    // gate = sigmoid(z@g_w) -> T1
    launch_gemm<ACT_SIG,  true>(ZL, CP, (const float*)d_in[base + T_G_W], CP,
                                (const float*)d_in[base + T_G_B], T1, CP, NPAIR, CP, CP, st);
    // zout = x_ln @ z_w + z_b -> T2
    launch_gemm<ACT_NONE, true>(T0, CP, (const float*)d_in[base + T_Z_W], CP,
                                (const float*)d_in[base + T_Z_B], T2, CP, NPAIR, CP, CP, st);
    // P1 += gate * zout
    fuse_gate_kernel<<<EW_BLK, 256, 0, st>>>(T1, T2, P1);
}

extern "C" void kernel_launch(void* const* d_in, const int* in_sizes, int n_in,
                              void* d_out, int out_size, void* d_ws, size_t ws_size,
                              hipStream_t stream) {
    (void)in_sizes; (void)n_in; (void)out_size; (void)ws_size;

    const float* s     = (const float*)d_in[IN_S];
    const float* p     = (const float*)d_in[IN_P];
    const float* trans = (const float*)d_in[IN_TRANS];
    const float* pmask = (const float*)d_in[IN_PMASK];

    // ---- workspace layout (256B aligned regions, explicit reuse) ----
    char* ws = (char*)d_ws;
    size_t off = 0;
    auto alloc = [&](size_t bytes) { size_t o = off; off += (bytes + 255) & ~(size_t)255; return o; };
    float*    P1 = (float*)   (ws + alloc((size_t)NPAIR * CP * 4)); // p accumulator
    _Float16* ZL = (_Float16*)(ws + alloc((size_t)NPAIR * CP * 2)); // LN output (f16)
    _Float16* T0 = (_Float16*)(ws + alloc((size_t)NPAIR * FK * 2)); // feats / h2 / gates / x_ln
    _Float16* T1 = (_Float16*)(ws + alloc((size_t)NPAIR * CP * 2)); // h1 / proj / gate
    _Float16* T2 = (_Float16*)(ws + alloc((size_t)NPAIR * CP * 2)); // upd / at / zout
    _Float16* T3 = (_Float16*)(ws + alloc((size_t)NPAIR * CP * 2)); // bt
    float*    XF = (float*)   (ws + alloc((size_t)NPAIR * CP * 4)); // einsum x / pt hidden (f16 N=256)
    float*    SB = (float*)   (ws + alloc((size_t)NRES * CP * 4));  // sb
    _Float16* SH = (_Float16*)(ws + alloc((size_t)NRES * CP * 2));  // s in f16

    const int EW_BLK = (NPAIR * CP) / (256 * 4);

    // ---- stage A: pair bias from single repr + RBF, symmetric MLP ----
    cvt_f32_f16_kernel<<<(NRES * CP) / 256, 256, 0, stream>>>(s, SH);
    launch_gemm<ACT_NONE, false>(SH, CP, (const float*)d_in[P_SA_W], CP,
                                 (const float*)d_in[P_SA_B], SB, CP, NRES, CP, CP, stream);
    pairfeat_kernel<<<NPAIR / 8, 256, 0, stream>>>(SB, trans,
                                                   (const float*)d_in[P_SALN_G],
                                                   (const float*)d_in[P_SALN_B], T0);
    launch_gemm<ACT_RELU, true>(T0, FK, (const float*)d_in[P_FA1_W], CP,
                                (const float*)d_in[P_FA1_B], T1, CP, NPAIR, CP, 148, stream);
    launch_gemm<ACT_RELU, true>(T1, CP, (const float*)d_in[P_FA2_W], CP,
                                (const float*)d_in[P_FA2_B], T0, CP, NPAIR, CP, CP, stream);
    launch_gemm<ACT_NONE, true>(T0, CP, (const float*)d_in[P_FA3_W], CP,
                                (const float*)d_in[P_FA3_B], T2, CP, NPAIR, CP, CP, stream);
    fuse_bias_kernel<<<EW_BLK, 256, 0, stream>>>(p, T2, s, P1);

    // ---- stage B: triangle multiplications (outgoing, incoming) ----
    run_tri_mul(d_in, TMO_BASE, true,  P1, ZL, T0, T1, T2, T3, XF, pmask, stream);
    run_tri_mul(d_in, TMI_BASE, false, P1, ZL, T0, T1, T2, T3, XF, pmask, stream);

    // ---- stage C: pair transition ----
    ln_kernel<<<NPAIR / 8, 256, 0, stream>>>(P1, (const float*)d_in[P_PT_LN_G],
                                             (const float*)d_in[P_PT_LN_B], ZL);
    _Float16* H = (_Float16*)XF;  // NPAIR x 256 f16 fits exactly in XF bytes
    launch_gemm<ACT_RELU, true>(ZL, CP, (const float*)d_in[P_PT1_W], 2 * CP,
                                (const float*)d_in[P_PT1_B], H, 2 * CP, NPAIR, 2 * CP, CP, stream);
    launch_gemm<ACT_NONE, true>(H, 2 * CP, (const float*)d_in[P_PT2_W], CP,
                                (const float*)d_in[P_PT2_B], T0, CP, NPAIR, CP, 2 * CP, stream);
    final_kernel<<<EW_BLK, 256, 0, stream>>>(P1, T0, pmask, (float*)d_out);
}
